// MyGRU_12060268167707
// MI455X (gfx1250) — matrix operand order repaired and checked by reading, not compiled, not run on hardware
//
#include <hip/hip_runtime.h>
#include <hip/hip_bf16.h>
#include <math.h>

// Problem dims
#define B_   128
#define T_   512
#define D_   256
#define U_   768
#define G3_  2304   // 3*U
#define NC_  16

typedef __bf16 v16bf __attribute__((ext_vector_type(16)));
typedef __bf16 v8bf  __attribute__((ext_vector_type(8)));
typedef float  v8f   __attribute__((ext_vector_type(8)));

union FragBf {
    v16bf v;
    struct { v8bf lo, hi; } p;
};

// ---------------------------------------------------------------------------
// Fragment loaders (wave32, 16x16x32 bf16 WMMA layouts per CDNA5 ISA 7.12.2)
// ---------------------------------------------------------------------------

// A-matrix 16x32 bf16, row-major source with leading dim `ld` (elements).
// lane&15 = row M; lane>>4 selects K-half: VGPR0..3 hold K = kh*8+0..7,
// VGPR4..7 hold K = 16+kh*8+0..7.
__device__ __forceinline__ v16bf load_a_bf16(const __bf16* A, int ld, int lane) {
    int m  = lane & 15;
    int kh = lane >> 4;
    const __bf16* p = A + (size_t)m * ld + kh * 8;
    FragBf f;
    f.p.lo = *(const v8bf*)(p);
    f.p.hi = *(const v8bf*)(p + 16);
    return f.v;
}

// B-matrix 32x16 bf16 from a TRANSPOSED weight (row-major N x K, ld = K).
// lane&15 = column N; lane>>4 selects K-half: VGPR0..3 hold K = kh*8+0..7,
// VGPR4..7 hold K = 16+kh*8+0..7 (matches A).
__device__ __forceinline__ v16bf load_b_bf16t(const __bf16* Bt, int ld, int lane) {
    int n  = lane & 15;
    int kh = lane >> 4;
    const __bf16* p = Bt + (size_t)n * ld + kh * 8;
    FragBf f;
    f.p.lo = *(const v8bf*)(p);
    f.p.hi = *(const v8bf*)(p + 16);
    return f.v;
}

__device__ __forceinline__ v8f wmma_bf16(v16bf a, v16bf b, v8f c) {
    return __builtin_amdgcn_wmma_f32_16x16x32_bf16(false, a, false, b, (short)0, c, false, false);
}

__device__ __forceinline__ float sigmoidf_(float x) {
    return 1.0f / (1.0f + __expf(-x));
}

// ---------------------------------------------------------------------------
// Utility kernels
// ---------------------------------------------------------------------------
__global__ void k_f32_to_bf16(const float* __restrict__ src, __bf16* __restrict__ dst, int n) {
    int i = blockIdx.x * blockDim.x + threadIdx.x;
    if (i < n) dst[i] = (__bf16)src[i];
}

// src: K x N row-major fp32  ->  dst: N x K row-major bf16 (transpose+convert)
__global__ void k_transpose_bf16(const float* __restrict__ src, __bf16* __restrict__ dst,
                                 int K, int N) {
    int i = blockIdx.x * blockDim.x + threadIdx.x;
    if (i >= K * N) return;
    int k = i / N;
    int n = i - k * N;
    dst[(size_t)n * K + k] = (__bf16)src[(size_t)k * N + n];
}

__global__ void k_zero_f32(float* __restrict__ p, int n) {
    int i = blockIdx.x * blockDim.x + threadIdx.x;
    if (i < n) p[i] = 0.0f;
}

__global__ void k_zero_bf16(__bf16* __restrict__ p, int n) {
    int i = blockIdx.x * blockDim.x + threadIdx.x;
    if (i < n) p[i] = (__bf16)0.0f;
}

// ---------------------------------------------------------------------------
// Input-projection GEMM:  C[M,N] = A[M,K](bf16) * Bt[N,K]^T(bf16) + bias[N]
// Block: 256 threads = 8 waves laid out 4(row) x 2(col); each wave computes a
// 32x32 tile (2 A-frags x 2 B-frags -> 4 WMMAs, 2x operand reuse).
// Block tile: 128 rows x 64 cols.
// ---------------------------------------------------------------------------
__global__ __launch_bounds__(256)
void k_gemm_xp(const __bf16* __restrict__ A, const __bf16* __restrict__ Bt,
               const float* __restrict__ bias, float* __restrict__ C,
               int M, int K, int N) {
    int wave = threadIdx.x >> 5;
    int lane = threadIdx.x & 31;
    int wr   = wave & 3;           // 4 row-waves
    int wc   = wave >> 2;          // 2 col-waves
    int row0 = blockIdx.y * 128 + wr * 32;
    int n0   = blockIdx.x * 64 + wc * 32;

    const __bf16* A0 = A  + (size_t)row0 * K;
    const __bf16* A1 = A0 + (size_t)16 * K;
    const __bf16* B0 = Bt + (size_t)n0 * K;
    const __bf16* B1 = B0 + (size_t)16 * K;

    v8f c00 = {}, c01 = {}, c10 = {}, c11 = {};
    for (int k0 = 0; k0 < K; k0 += 32) {
        __builtin_prefetch(A0 + k0 + 256, 0, 1);
        __builtin_prefetch(A1 + k0 + 256, 0, 1);
        v16bf a0 = load_a_bf16(A0 + k0, K, lane);
        v16bf a1 = load_a_bf16(A1 + k0, K, lane);
        v16bf b0 = load_b_bf16t(B0 + k0, K, lane);
        v16bf b1 = load_b_bf16t(B1 + k0, K, lane);
        c00 = wmma_bf16(a0, b0, c00);
        c01 = wmma_bf16(a0, b1, c01);
        c10 = wmma_bf16(a1, b0, c10);
        c11 = wmma_bf16(a1, b1, c11);
    }

    int cl  = lane & 15;
    int rhi = (lane >> 4) << 3;
    float bc0 = bias[n0 + cl];
    float bc1 = bias[n0 + 16 + cl];
#pragma unroll
    for (int j = 0; j < 8; ++j) {
        int r0 = row0 + j + rhi;
        int r1 = r0 + 16;
        C[(size_t)r0 * N + n0 + cl]      = c00[j] + bc0;
        C[(size_t)r0 * N + n0 + 16 + cl] = c01[j] + bc1;
        C[(size_t)r1 * N + n0 + cl]      = c10[j] + bc0;
        C[(size_t)r1 * N + n0 + 16 + cl] = c11[j] + bc1;
    }
}

// ---------------------------------------------------------------------------
// One GRU time step (reset_after=True, gate order z,r,h):
//   rec = h_in @ U + br ; z=sig(xz+rz); r=sig(xr+rr); hh=tanh(xh+r*rh)
//   h_out = z*h_in + (1-z)*hh
// Block: 256 threads = 8 waves = 4 u-tiles x 2 K-halves. Each wave runs the
// three gate WMMA chains over K = 384; the upper-K wave dumps its partials to
// LDS, lower-K wave reduces and fuses the gate nonlinearities.
// Hidden state kept in fp32 (hinf) AND bf16 (hinb, WMMA A operand).
// xp rows are indexed (b*T + t); seq (optional) likewise.
// ---------------------------------------------------------------------------
__global__ __launch_bounds__(256)
void k_gru_step(const float* __restrict__ hinf, const __bf16* __restrict__ hinb,
                float* __restrict__ houtf, __bf16* __restrict__ houtb,
                const __bf16* __restrict__ Ut,      // [3U x U] bf16 (transposed)
                const float* __restrict__ br,       // recurrent bias [3U]
                const float* __restrict__ xp,       // [B*T x 3U] fp32
                __bf16* __restrict__ seq,           // [B*T x U] bf16 or nullptr
                int t) {
    __shared__ float red[4 * 3 * 32 * 8];           // 12 KB partial accumulators

    int wave  = threadIdx.x >> 5;
    int lane  = threadIdx.x & 31;
    int utile = wave & 3;
    int khalf = wave >> 2;                          // 0: K=[0,384), 1: K=[384,768)
    int n0    = (blockIdx.x * 4 + utile) * 16;      // u-column tile
    int row0  = blockIdx.y * 16;                    // batch-row tile
    int kbase = khalf * (U_ / 2);

    const __bf16* Arow = hinb + (size_t)row0 * U_ + kbase;
    const __bf16* Bz = Ut + (size_t)(0 * U_ + n0) * U_ + kbase;
    const __bf16* Br = Ut + (size_t)(1 * U_ + n0) * U_ + kbase;
    const __bf16* Bh = Ut + (size_t)(2 * U_ + n0) * U_ + kbase;

    v8f cz = {}, cr = {}, ch = {};
    for (int k0 = 0; k0 < U_ / 2; k0 += 32) {
        v16bf af = load_a_bf16(Arow + k0, U_, lane);
        cz = wmma_bf16(af, load_b_bf16t(Bz + k0, U_, lane), cz);
        cr = wmma_bf16(af, load_b_bf16t(Br + k0, U_, lane), cr);
        ch = wmma_bf16(af, load_b_bf16t(Bh + k0, U_, lane), ch);
    }

    if (khalf == 1) {
        int base = ((utile * 3 + 0) * 32 + lane) * 8;
        *(v8f*)(red + base)            = cz;
        *(v8f*)(red + base + 32 * 8)   = cr;
        *(v8f*)(red + base + 64 * 8)   = ch;
    }
    __syncthreads();
    if (khalf == 0) {
        int base = ((utile * 3 + 0) * 32 + lane) * 8;
        cz += *(const v8f*)(red + base);
        cr += *(const v8f*)(red + base + 32 * 8);
        ch += *(const v8f*)(red + base + 64 * 8);

        int col = n0 + (lane & 15);
        int rhi = (lane >> 4) << 3;
        float bz = br[col], brr = br[U_ + col], bh = br[2 * U_ + col];

#pragma unroll
        for (int j = 0; j < 8; ++j) {
            int row = row0 + j + rhi;               // batch index
            size_t xrow = ((size_t)row * T_ + t) * G3_;
            float xz = xp[xrow + col];
            float xr = xp[xrow + U_ + col];
            float xh = xp[xrow + 2 * U_ + col];
            float rz = cz[j] + bz;
            float rr = cr[j] + brr;
            float rh = ch[j] + bh;
            float z  = sigmoidf_(xz + rz);
            float r  = sigmoidf_(xr + rr);
            float hh = tanhf(xh + r * rh);
            float ho = hinf[(size_t)row * U_ + col];
            float hn = z * ho + (1.0f - z) * hh;
            houtf[(size_t)row * U_ + col] = hn;
            houtb[(size_t)row * U_ + col] = (__bf16)hn;
            if (seq) seq[((size_t)row * T_ + t) * U_ + col] = (__bf16)hn;
        }
    }
}

// ---------------------------------------------------------------------------
// Head: logits = h2 @ Wo + bo ; softmax over 16 classes. One thread per row.
// ---------------------------------------------------------------------------
__global__ void k_head(const float* __restrict__ h2, const float* __restrict__ Wo,
                       const float* __restrict__ bo, float* __restrict__ out) {
    int row = threadIdx.x;
    if (row >= B_) return;
    float l[NC_];
#pragma unroll
    for (int c = 0; c < NC_; ++c) l[c] = bo[c];
    for (int k = 0; k < U_; ++k) {
        float hv = h2[(size_t)row * U_ + k];
#pragma unroll
        for (int c = 0; c < NC_; ++c) l[c] += hv * Wo[(size_t)k * NC_ + c];
    }
    float mx = l[0];
#pragma unroll
    for (int c = 1; c < NC_; ++c) mx = fmaxf(mx, l[c]);
    float sum = 0.0f;
#pragma unroll
    for (int c = 0; c < NC_; ++c) { l[c] = __expf(l[c] - mx); sum += l[c]; }
    float inv = 1.0f / sum;
#pragma unroll
    for (int c = 0; c < NC_; ++c) out[(size_t)row * NC_ + c] = l[c] * inv;
}

// ---------------------------------------------------------------------------
// Host orchestration
// ---------------------------------------------------------------------------
static inline size_t align256(size_t x) { return (x + 255) & ~(size_t)255; }

extern "C" void kernel_launch(void* const* d_in, const int* in_sizes, int n_in,
                              void* d_out, int out_size, void* d_ws, size_t ws_size,
                              hipStream_t stream) {
    (void)in_sizes; (void)n_in; (void)out_size; (void)ws_size;
    const float* x   = (const float*)d_in[0];   // [B,T,D]
    const float* W1  = (const float*)d_in[1];   // [D,3U]
    const float* U1  = (const float*)d_in[2];   // [U,3U]
    const float* b1  = (const float*)d_in[3];   // [2,3U]
    const float* W2  = (const float*)d_in[4];   // [U,3U]
    const float* U2  = (const float*)d_in[5];   // [U,3U]
    const float* b2  = (const float*)d_in[6];   // [2,3U]
    const float* Wo  = (const float*)d_in[7];   // [U,NC]
    const float* bo  = (const float*)d_in[8];   // [NC]
    float* out = (float*)d_out;

    // Workspace carve-up
    char* ws = (char*)d_ws;
    size_t off = 0;
    __bf16* xbf   = (__bf16*)(ws + off); off = align256(off + (size_t)B_ * T_ * D_ * 2);
    __bf16* w1t   = (__bf16*)(ws + off); off = align256(off + (size_t)G3_ * D_ * 2);
    __bf16* u1t   = (__bf16*)(ws + off); off = align256(off + (size_t)G3_ * U_ * 2);
    __bf16* w2t   = (__bf16*)(ws + off); off = align256(off + (size_t)G3_ * U_ * 2);
    __bf16* u2t   = (__bf16*)(ws + off); off = align256(off + (size_t)G3_ * U_ * 2);
    __bf16* seqb  = (__bf16*)(ws + off); off = align256(off + (size_t)B_ * T_ * U_ * 2);
    float*  ha    = (float*)(ws + off);  off = align256(off + (size_t)B_ * U_ * 4);
    float*  hb    = (float*)(ws + off);  off = align256(off + (size_t)B_ * U_ * 4);
    __bf16* habf  = (__bf16*)(ws + off); off = align256(off + (size_t)B_ * U_ * 2);
    __bf16* hbbf  = (__bf16*)(ws + off); off = align256(off + (size_t)B_ * U_ * 2);
    float*  xp    = (float*)(ws + off);  off = align256(off + (size_t)B_ * T_ * G3_ * 4);

    // 1. Convert inputs + weights to bf16 (weights transposed to [N x K])
    {
        int n = B_ * T_ * D_;
        k_f32_to_bf16<<<(n + 255) / 256, 256, 0, stream>>>(x, xbf, n);
    }
    k_transpose_bf16<<<(D_ * G3_ + 255) / 256, 256, 0, stream>>>(W1, w1t, D_, G3_);
    k_transpose_bf16<<<(U_ * G3_ + 255) / 256, 256, 0, stream>>>(U1, u1t, U_, G3_);
    k_transpose_bf16<<<(U_ * G3_ + 255) / 256, 256, 0, stream>>>(W2, w2t, U_, G3_);
    k_transpose_bf16<<<(U_ * G3_ + 255) / 256, 256, 0, stream>>>(U2, u2t, U_, G3_);
    k_zero_f32<<<(B_ * U_ + 255) / 256, 256, 0, stream>>>(ha, B_ * U_);
    k_zero_bf16<<<(B_ * U_ + 255) / 256, 256, 0, stream>>>(habf, B_ * U_);

    const int M = B_ * T_;
    dim3 gGemm(G3_ / 64, M / 128);
    dim3 gStep(U_ / 16 / 4, B_ / 16);

    // 2. Layer-1 input projection: xp = xbf @ W1 + b1[0]
    k_gemm_xp<<<gGemm, 256, 0, stream>>>(xbf, w1t, b1, xp, M, D_, G3_);

    // 3. Layer-1 recurrence (ping-pong h fp32+bf16; seq written for layer 2)
    for (int t = 0; t < T_; ++t) {
        const float*  hi  = (t & 1) ? hb : ha;
        float*        ho  = (t & 1) ? ha : hb;
        const __bf16* hbi = (t & 1) ? hbbf : habf;
        __bf16*       hbo = (t & 1) ? habf : hbbf;
        k_gru_step<<<gStep, 256, 0, stream>>>(hi, hbi, ho, hbo, u1t, b1 + G3_, xp, seqb, t);
    }
    // T_=512 even -> final h1 lands in ha / habf.

    // 4. Layer-2 input projection: xp = seqb @ W2 + b2[0]
    k_gemm_xp<<<gGemm, 256, 0, stream>>>(seqb, w2t, b2, xp, M, U_, G3_);

    // 5. Layer-2 recurrence, h0 = h1 (already in ha/habf); no seq output.
    for (int t = 0; t < T_; ++t) {
        const float*  hi  = (t & 1) ? hb : ha;
        float*        ho  = (t & 1) ? ha : hb;
        const __bf16* hbi = (t & 1) ? hbbf : habf;
        __bf16*       hbo = (t & 1) ? habf : hbbf;
        k_gru_step<<<gStep, 256, 0, stream>>>(hi, hbi, ho, hbo, u2t, b2 + G3_, xp, (__bf16*)nullptr, t);
    }
    // final h2 in ha.

    // 6. Dense head + softmax
    k_head<<<1, 128, 0, stream>>>(ha, Wo, bo, out);
}
